// GraphConvolution_13211319403105
// MI455X (gfx1250) — compile-verified, matrix operand order checked by reading, not run on hardware
//
#include <hip/hip_runtime.h>
#include <hip/hip_bf16.h>

#define N_NODES 100000
#define N_EDGES 3200000
#define D_IN    512
#define D_OUT   256

typedef __attribute__((ext_vector_type(16))) __bf16 v16bf;
typedef __attribute__((ext_vector_type(8)))  __bf16 v8bf;
typedef __attribute__((ext_vector_type(8)))  float  v8f;
typedef __attribute__((ext_vector_type(4)))  float  v4f;
typedef __attribute__((ext_vector_type(4)))  int    v4i;

typedef __attribute__((address_space(1))) v4i* global_v4i_ptr;
typedef __attribute__((address_space(3))) v4i* shared_v4i_ptr;

#define KCHUNK  32                 // one WMMA K-step per LDS chunk
#define NCHUNKS (D_IN / KCHUNK)    // 16
#define LDS_K   40                 // 32 + 8 pad (stride 80B: spreads banks)

#if __has_builtin(__builtin_amdgcn_global_load_async_to_lds_b128) && \
    __has_builtin(__builtin_amdgcn_s_wait_asynccnt)
#define USE_ASYNC_LDS 1
#else
#define USE_ASYNC_LDS 0
#endif

// f32 -> bf16 RNE via integer ops (no reliance on __bf16 cvt codegen)
__device__ __forceinline__ __bf16 f2bf(float f) {
    unsigned u = __builtin_bit_cast(unsigned, f);
    unsigned r = (u + 0x7FFFu + ((u >> 16) & 1u)) >> 16;
    unsigned short s = (unsigned short)r;
    return __builtin_bit_cast(__bf16, s);
}

// -------------------------------------------------------------------------
// One-time: Wt[n][k] = bf16(W[k][n])  (256 x 512 bf16 = 256 KB, L2-resident)
// -------------------------------------------------------------------------
__global__ __launch_bounds__(256)
void gcn_prep_wt(const float* __restrict__ W, __bf16* __restrict__ Wt)
{
    const int n = blockIdx.x;                       // 0..255
    for (int k = threadIdx.x; k < D_IN; k += 256)
        Wt[(size_t)n * D_IN + k] = f2bf(W[(size_t)k * D_OUT + n]);
}

// -------------------------------------------------------------------------
// support = bf16(X) @ W_bf16, fp32 accumulate, WMMA 16x16x32.
// Block = 256 threads = 8 waves; wave owns a 16-row M-tile x 256 cols.
// W K-chunks staged to double-buffered LDS via async global->LDS copies,
// overlapped with WMMA on the other buffer. A raw loads software-pipelined.
// -------------------------------------------------------------------------
__global__ __launch_bounds__(256)
void gcn_gemm_bf16_wmma(const float* __restrict__ X,
                        const __bf16* __restrict__ Wt,   // [D_OUT][D_IN]
                        float* __restrict__ support)
{
    __shared__ __bf16 ldsW[2][D_OUT * LDS_K];            // 2 x 20,480 B

    const int tid  = threadIdx.x;
    const int lane = tid & 31;
    const int wave = tid >> 5;
    const int rowBase = blockIdx.x * 128 + wave * 16;

    // A layout: lanes 0-15: M=lane, K {0..7,16..23}; lanes 16-31: M=lane-16, K {8..15,24..31}
    const int mRow = rowBase + (lane & 15);
    const int aRow = (mRow < N_NODES) ? mRow : (N_NODES - 1);  // clamp: EXEC all-1s
    const int kSel = (lane >> 4) * 8;

    v8f acc[16] = {};

    // Stage chunk -> LDS buf: thread t owns column n=t, 4 x 16B transfers.
    auto stage = [&](int chunk, int buf) {
        const __bf16* src = Wt + (size_t)tid * D_IN + chunk * KCHUNK;
        __bf16* dst = &ldsW[buf][tid * LDS_K];
#if USE_ASYNC_LDS
        #pragma unroll
        for (int kb = 0; kb < 4; ++kb) {
            __builtin_amdgcn_global_load_async_to_lds_b128(
                (global_v4i_ptr)(src + kb * 8),
                (shared_v4i_ptr)(dst + kb * 8),
                0, 0);
        }
#else
        #pragma unroll
        for (int kb = 0; kb < 4; ++kb)
            *(v8bf*)(dst + kb * 8) = *(const v8bf*)(src + kb * 8);
#endif
    };

    stage(0, 0);

    // A raw prefetch (chunk 0)
    const float* aBase = X + (size_t)aRow * D_IN + kSel;
    v4f ar0 = *(const v4f*)(aBase + 0);
    v4f ar1 = *(const v4f*)(aBase + 4);
    v4f ar2 = *(const v4f*)(aBase + 16);
    v4f ar3 = *(const v4f*)(aBase + 20);

    for (int c = 0; c < NCHUNKS; ++c) {
        const int buf = c & 1;
        v4f an0 = {}, an1 = {}, an2 = {}, an3 = {};

        if (c + 1 < NCHUNKS) {
            stage(c + 1, buf ^ 1);                       // async fill other buffer
            const float* ap = aBase + (c + 1) * KCHUNK;  // prefetch next A raw
            an0 = *(const v4f*)(ap + 0);
            an1 = *(const v4f*)(ap + 4);
            an2 = *(const v4f*)(ap + 16);
            an3 = *(const v4f*)(ap + 20);
#if USE_ASYNC_LDS
            __builtin_amdgcn_s_wait_asynccnt(4);         // chunk c's 4 done; next 4 in flight
#endif
        } else {
#if USE_ASYNC_LDS
            __builtin_amdgcn_s_wait_asynccnt(0);
#endif
        }
        __syncthreads();                                 // chunk c visible from all waves

        // Build A fragment (16 bf16)
        v16bf A;
        #pragma unroll
        for (int j = 0; j < 4; ++j) {
            A[j]      = f2bf(ar0[j]);
            A[4 + j]  = f2bf(ar1[j]);
            A[8 + j]  = f2bf(ar2[j]);
            A[12 + j] = f2bf(ar3[j]);
        }

        // 16 N-tiles, B fragments double-buffered in registers
        const int nColBase = lane & 15;
        const __bf16* lbase = &ldsW[buf][0];
        v8bf b0[2], b1[2];
        {
            const __bf16* bp = lbase + nColBase * LDS_K + kSel;
            b0[0] = *(const v8bf*)(bp);
            b1[0] = *(const v8bf*)(bp + 16);
        }
        #pragma unroll
        for (int nt = 0; nt < 16; ++nt) {
            if (nt + 1 < 16) {
                const __bf16* bp = lbase + ((nt + 1) * 16 + nColBase) * LDS_K + kSel;
                b0[(nt + 1) & 1] = *(const v8bf*)(bp);
                b1[(nt + 1) & 1] = *(const v8bf*)(bp + 16);
            }
            v16bf B;
            #pragma unroll
            for (int j = 0; j < 8; ++j) { B[j] = b0[nt & 1][j]; B[8 + j] = b1[nt & 1][j]; }
            acc[nt] = __builtin_amdgcn_wmma_f32_16x16x32_bf16(
                false, A, false, B, (short)0, acc[nt], false, false);
        }

        ar0 = an0; ar1 = an1; ar2 = an2; ar3 = an3;
        __syncthreads();                                 // all reads of buf done before refill
    }

    // C/D layout: lanes 0-15 -> N=lane, VGPR r -> M=r; lanes 16-31 -> M=r+8
    const int nCol = lane & 15;
    const int rOff = (lane >> 4) * 8;
    #pragma unroll
    for (int nt = 0; nt < 16; ++nt) {
        #pragma unroll
        for (int r = 0; r < 8; ++r) {
            const int row = rowBase + r + rOff;
            if (row < N_NODES)
                support[(size_t)row * D_OUT + nt * 16 + nCol] = acc[nt][r];
        }
    }
}

// -------------------------------------------------------------------------
// out[r][n] = bias[n]  (out then accumulated atomically)
// -------------------------------------------------------------------------
__global__ void gcn_init_out(const float* __restrict__ bias, float* __restrict__ out)
{
    size_t i = (size_t)blockIdx.x * blockDim.x + threadIdx.x;
    if (i < (size_t)N_NODES * D_OUT)
        out[i] = bias[i & (D_OUT - 1)];
}

// -------------------------------------------------------------------------
// COO scatter: out[r][:] += v * support[c][:]. One wave per edge; lane owns
// 8 contiguous cols. support & out are L2-resident (102.4 MB < 192 MB L2).
// -------------------------------------------------------------------------
__global__ __launch_bounds__(256)
void gcn_scatter(const int* __restrict__ rows, const int* __restrict__ cols,
                 const float* __restrict__ vals, const float* __restrict__ support,
                 float* __restrict__ out)
{
    const int lane = threadIdx.x & 31;
    const int wave = threadIdx.x >> 5;
    const int e = blockIdx.x * 8 + wave;
    if (e >= N_EDGES) return;

    const int   r = rows[e];
    const int   c = cols[e];
    const float v = vals[e];

    const float* src = support + (size_t)c * D_OUT + lane * 8;
    float*       dst = out     + (size_t)r * D_OUT + lane * 8;

    v4f s0 = *(const v4f*)(src);
    v4f s1 = *(const v4f*)(src + 4);
    #pragma unroll
    for (int j = 0; j < 4; ++j) {
        atomicAdd(dst + j,     v * s0[j]);
        atomicAdd(dst + 4 + j, v * s1[j]);
    }
}

extern "C" void kernel_launch(void* const* d_in, const int* in_sizes, int n_in,
                              void* d_out, int out_size, void* d_ws, size_t ws_size,
                              hipStream_t stream)
{
    const int*   adj_row = (const int*)  d_in[0];
    const int*   adj_col = (const int*)  d_in[1];
    const float* adj_val = (const float*)d_in[2];
    const float* X       = (const float*)d_in[3];
    const float* W       = (const float*)d_in[4];
    const float* bias    = (const float*)d_in[5];
    float* out = (float*)d_out;

    // Workspace: [support: 100000*256 f32 = 102,400,000 B][Wt: 256*512 bf16 = 262,144 B]
    float*  support = (float*)d_ws;
    __bf16* Wt      = (__bf16*)((char*)d_ws + (size_t)N_NODES * D_OUT * sizeof(float));

    // 1) Wt = bf16(W^T)
    gcn_prep_wt<<<D_OUT, 256, 0, stream>>>(W, Wt);

    // 2) support = X @ W  (bf16 WMMA, fp32 accum)
    gcn_gemm_bf16_wmma<<<(N_NODES + 127) / 128, 256, 0, stream>>>(X, Wt, support);

    // 3) out = broadcast(bias)
    const size_t total = (size_t)N_NODES * D_OUT;
    gcn_init_out<<<(unsigned)((total + 255) / 256), 256, 0, stream>>>(bias, out);

    // 4) out += A_sparse @ support
    gcn_scatter<<<(N_EDGES + 7) / 8, 256, 0, stream>>>(adj_row, adj_col, adj_val,
                                                       support, out);
}
